// AdvancedTransactionGNN_41051297415257
// MI455X (gfx1250) — matrix operand order repaired and checked by reading, not compiled, not run on hardware
//
#include <hip/hip_runtime.h>
#include <hip/hip_bf16.h>
#include <cstdint>

// ---------------------------------------------------------------------------
// CDNA5 (gfx1250) GAT implementation.
// GEMMs: pre-converted bf16 operands, async global->LDS double-buffered B
// panel (GLOBAL_LOAD_ASYNC_TO_LDS_B128 + s_wait_asynccnt), and
// v_wmma_f32_16x16x32_bf16 with f32 accumulation.
// ---------------------------------------------------------------------------

typedef __attribute__((ext_vector_type(16))) __bf16 v16bf;
typedef __attribute__((ext_vector_type(8)))  __bf16 v8bf;
typedef __attribute__((ext_vector_type(8)))  float  v8f;

__device__ __forceinline__ __bf16 f2bf(float f) {
    // round-to-nearest-even f32 -> bf16
    unsigned u = __float_as_uint(f);
    unsigned r = (u + 0x7FFFu + ((u >> 16) & 1u)) >> 16;
    unsigned short hs = (unsigned short)r;
    __bf16 out;
    __builtin_memcpy(&out, &hs, 2);
    return out;
}

// Elementwise f32 -> bf16 conversion (activations), done once per layer.
__global__ __launch_bounds__(256) void conv_bf16(
    const float* __restrict__ in, __bf16* __restrict__ out, size_t n)
{
    size_t i = blockIdx.x * (size_t)blockDim.x + threadIdx.x;
    if (i < n) out[i] = f2bf(in[i]);
}

// Pack W[K,M] (f32 row-major) into WMMA B-fragment order, bf16:
//   Wp[colTile][kTile][lane][e] , colTile = M/16, kTile = K/32.
//   lane 0-15 : col = lane,    e -> K = kTile*32 + e
//   lane 16-31: col = lane-16, e -> K = kTile*32 + 16 + e
// One thread per (colTile,kTile,lane): writes 16 contiguous bf16 (32 bytes).
__global__ __launch_bounds__(256) void pack_W(
    const float* __restrict__ W, __bf16* __restrict__ Wp, int K, int M)
{
    int tid = (int)(blockIdx.x * (size_t)blockDim.x + threadIdx.x);
    int nkt = K >> 5;
    int total = (M >> 4) * nkt * 32;
    if (tid >= total) return;
    int lane = tid & 31;
    int kt   = (tid >> 5) % nkt;
    int ct   = (tid >> 5) / nkt;
    int col   = ct * 16 + (lane & 15);
    int kbase = kt * 32 + (lane >> 4) * 16;
    __bf16* out = Wp + (size_t)tid * 16;
#pragma unroll
    for (int e = 0; e < 16; ++e) out[e] = f2bf(W[(size_t)(kbase + e) * M + col]);
}

// C[N,M] = A[N,K] @ W[K,M]; A bf16 row-major, Wp packed bf16 fragments, C f32.
// Block: 256 threads = 8 waves; each wave: 16 rows x (TN*16) cols; all waves
// share the same column tiles, so the B panel is staged into LDS once per
// block with async loads, double-buffered across k-tiles.
// grid.x = M/(TN*16), grid.y = ceil(N/128). Requires K%32==0, N%16==0.
template <int TN>
__global__ __launch_bounds__(256) void gemm_bf16_wmma(
    const __bf16* __restrict__ A, const __bf16* __restrict__ Wp,
    float* __restrict__ C, int N, int K, int M)
{
    __shared__ __align__(16) unsigned char ldsB[2 * TN * 1024];

    const int wave = threadIdx.x >> 5;
    const int lane = threadIdx.x & 31;
    const int row0 = blockIdx.y * 128 + wave * 16;
    const int col0 = blockIdx.x * (TN * 16);
    const int nkt  = K >> 5;

    // Clamp OOB-row waves to a valid row so every wave can reach the barriers;
    // their stores are predicated off below. (N % 16 == 0 by construction.)
    const int rowA   = (row0 < N) ? row0 : (N - 16);
    const bool valid = (row0 < N);

    v8f zero = {};
    v8f acc[TN];
#pragma unroll
    for (int t = 0; t < TN; ++t) acc[t] = zero;

    // A fragment (16-bit A 16x32, contiguous k order, same as pack_W):
    //   lanes 0-15 : row=lane,    elems[0..15]=K 0..15
    //   lanes 16-31: row=lane-16, elems[0..15]=K 16..31
    const __bf16* arp = A + (size_t)(rowA + (lane & 15)) * K + (lane >> 4) * 16;

    // B staging: thread (wave w < TN, lane l) owns tile t=w's 32-byte lane
    // chunk. Global source for k-tile kt: Wp + ((ct0+w)*nkt + kt)*1024 + l*32.
    const bool stager = (wave < TN);
    const unsigned ldsBase = (unsigned)(size_t)&ldsB[0];   // LDS addr = addr[31:0]
    const char* gbase = (const char*)Wp +
        (size_t)(blockIdx.x * TN + wave) * nkt * 1024 + lane * 32;

    // Prologue: stage k-tile 0 into buffer 0.
    if (stager) {
        unsigned d = ldsBase + wave * 1024 + lane * 32;
        unsigned long long g = (unsigned long long)(size_t)gbase;
        asm volatile(
            "global_load_async_to_lds_b128 %0, %1, off\n\t"
            "global_load_async_to_lds_b128 %0, %1, off offset:16"
            :: "v"(d), "v"(g) : "memory");
    }

    for (int kt = 0; kt < nkt; ++kt) {
        const int buf = kt & 1;
        // Issue next stage into the other buffer before waiting.
        if (kt + 1 < nkt) {
            if (stager) {
                unsigned d = ldsBase + ((kt + 1) & 1) * (TN * 1024) + wave * 1024 + lane * 32;
                unsigned long long g =
                    (unsigned long long)(size_t)(gbase + (size_t)(kt + 1) * 1024);
                asm volatile(
                    "global_load_async_to_lds_b128 %0, %1, off\n\t"
                    "global_load_async_to_lds_b128 %0, %1, off offset:16"
                    :: "v"(d), "v"(g) : "memory");
            }
            // Each stage = 2 async loads/thread; allow the 2 just-issued
            // next-stage loads to stay outstanding (in-order completion).
            asm volatile("s_wait_asynccnt 0x2" ::: "memory");
        } else {
            asm volatile("s_wait_asynccnt 0x0" ::: "memory");
        }
        __syncthreads();

        // A fragment for this k-tile (per-wave unique; direct b128 loads).
        v8bf alo = *(const v8bf*)(arp + kt * 32);
        v8bf ahi = *(const v8bf*)(arp + kt * 32 + 8);
        v16bf a = __builtin_shufflevector(alo, ahi,
                                          0, 1, 2, 3, 4, 5, 6, 7,
                                          8, 9, 10, 11, 12, 13, 14, 15);

        const unsigned char* rb = &ldsB[buf * (TN * 1024)];
#pragma unroll
        for (int t = 0; t < TN; ++t) {
            v16bf b = *(const v16bf*)(rb + t * 1024 + lane * 32);
            acc[t] = __builtin_amdgcn_wmma_f32_16x16x32_bf16(
                false, a, false, b, (short)0, acc[t], false, false);
        }
        __syncthreads();   // all waves done reading `buf` before it is restaged
    }

    if (!valid) return;
    // C/D layout: VGPR r -> lanes 0-15: M=r, lanes 16-31: M=8+r; col = lane&15
    const int rbase = (lane >> 4) * 8;
    const int ccol  = lane & 15;
#pragma unroll
    for (int r = 0; r < 8; ++r) {
        float* cp = C + (size_t)(row0 + rbase + r) * M + col0 + ccol;
#pragma unroll
        for (int t = 0; t < TN; ++t) cp[t * 16] = acc[t][r];
    }
}

// alpha_s[n,h] = sum_d H[n,h,d]*a_src[h,d] ; alpha_d likewise. One wave per (n,h).
__global__ __launch_bounds__(256) void attn_coeff(
    const float* __restrict__ Hm, const float* __restrict__ a_src,
    const float* __restrict__ a_dst, float* __restrict__ alpha_s,
    float* __restrict__ alpha_d, int N, int Hh, int D)
{
    int wid  = (int)((blockIdx.x * (size_t)blockDim.x + threadIdx.x) >> 5);
    int lane = threadIdx.x & 31;
    if (wid >= N * Hh) return;
    int n = wid / Hh, h = wid - n * Hh;
    const float* hp = Hm + (size_t)n * Hh * D + (size_t)h * D;
    const float* as = a_src + (size_t)h * D;
    const float* ad = a_dst + (size_t)h * D;
    float ss = 0.f, sd = 0.f;
    for (int d = lane; d < D; d += 32) {
        float v = hp[d];
        ss += v * as[d];
        sd += v * ad[d];
    }
#pragma unroll
    for (int off = 16; off > 0; off >>= 1) {
        ss += __shfl_down(ss, off, 32);
        sd += __shfl_down(sd, off, 32);
    }
    if (lane == 0) { alpha_s[wid] = ss; alpha_d[wid] = sd; }
}

__device__ __forceinline__ unsigned mono_enc(float f) {
    unsigned u = __float_as_uint(f);
    return (u & 0x80000000u) ? ~u : (u | 0x80000000u);
}
__device__ __forceinline__ float mono_dec(unsigned u) {
    return (u & 0x80000000u) ? __uint_as_float(u ^ 0x80000000u) : __uint_as_float(~u);
}

// Pass 1: e = leaky_relu(as[src]+ad[dst]); m[dst] = max via monotone-uint atomicMax.
__global__ __launch_bounds__(256) void edge_max(
    const int* __restrict__ ei, int E, int N, int Hh,
    const float* __restrict__ as, const float* __restrict__ ad,
    unsigned int* __restrict__ m)
{
    int e = (int)(blockIdx.x * (size_t)blockDim.x + threadIdx.x);
    int Et = E + N;
    if (e >= Et) return;
    int src = (e < E) ? ei[e]     : (e - E);
    int dst = (e < E) ? ei[E + e] : (e - E);
    for (int h = 0; h < Hh; ++h) {
        float v = as[(size_t)src * Hh + h] + ad[(size_t)dst * Hh + h];
        v = v > 0.f ? v : 0.2f * v;
        atomicMax(m + (size_t)dst * Hh + h, mono_enc(v));
    }
}

// Pass 2: s[dst] += exp(e - m[dst])
__global__ __launch_bounds__(256) void edge_sum(
    const int* __restrict__ ei, int E, int N, int Hh,
    const float* __restrict__ as, const float* __restrict__ ad,
    const unsigned int* __restrict__ m, float* __restrict__ s)
{
    int e = (int)(blockIdx.x * (size_t)blockDim.x + threadIdx.x);
    int Et = E + N;
    if (e >= Et) return;
    int src = (e < E) ? ei[e]     : (e - E);
    int dst = (e < E) ? ei[E + e] : (e - E);
    for (int h = 0; h < Hh; ++h) {
        float v = as[(size_t)src * Hh + h] + ad[(size_t)dst * Hh + h];
        v = v > 0.f ? v : 0.2f * v;
        float mv = mono_dec(m[(size_t)dst * Hh + h]);
        atomicAdd(s + (size_t)dst * Hh + h, __expf(v - mv));
    }
}

// Pass 3: out[dst,h,:] += alpha * H[src,h,:], one wave per edge, lanes stride D.
__global__ __launch_bounds__(256) void edge_agg(
    const int* __restrict__ ei, int E, int N, int Hh, int D,
    const float* __restrict__ as, const float* __restrict__ ad,
    const unsigned int* __restrict__ m, const float* __restrict__ s,
    const float* __restrict__ Hm, float* __restrict__ out)
{
    int wid  = (int)((blockIdx.x * (size_t)blockDim.x + threadIdx.x) >> 5);
    int lane = threadIdx.x & 31;
    int Et = E + N;
    if (wid >= Et) return;
    int src = (wid < E) ? ei[wid]     : (wid - E);
    int dst = (wid < E) ? ei[E + wid] : (wid - E);
    for (int h = 0; h < Hh; ++h) {
        float v = as[(size_t)src * Hh + h] + ad[(size_t)dst * Hh + h];
        v = v > 0.f ? v : 0.2f * v;
        float mv = mono_dec(m[(size_t)dst * Hh + h]);
        float alpha = __expf(v - mv) / (s[(size_t)dst * Hh + h] + 1e-16f);
        const float* hp = Hm  + (size_t)src * Hh * D + (size_t)h * D;
        float*       op = out + (size_t)dst * Hh * D + (size_t)h * D;
        for (int d = lane; d < D; d += 32)
            atomicAdd(op + d, alpha * hp[d]);
    }
}

// x[n,f] = act(x[n,f] + b[f]); act = ELU when do_elu.
__global__ __launch_bounds__(256) void bias_act(
    float* __restrict__ x, const float* __restrict__ b, size_t total, int F, int do_elu)
{
    size_t i = blockIdx.x * (size_t)blockDim.x + threadIdx.x;
    if (i >= total) return;
    float v = x[i] + b[i % (size_t)F];
    if (do_elu) v = v > 0.f ? v : (__expf(v) - 1.f);
    x[i] = v;
}

// ---------------------------------------------------------------------------

static inline size_t alignUp(size_t v, size_t a) { return (v + a - 1) & ~(a - 1); }

static void run_layer(const float* x_in, const float* W, const float* a_s, const float* a_d,
                      const float* b, float* Hbuf, float* outBuf, __bf16* Abf, __bf16* Wp,
                      float* alpha_s, float* alpha_d, unsigned* mbuf, float* sbuf,
                      const int* ei, int N, int E, int K, int Hh, int D, int do_elu,
                      hipStream_t stream)
{
    const int M = Hh * D;              // output feature width
    const int Et = E + N;

    // bf16 conversions: activations (row-major) and weights (fragment-packed)
    size_t nA = (size_t)N * K;
    conv_bf16<<<(int)((nA + 255) / 256), 256, 0, stream>>>(x_in, Abf, nA);
    int nWfrag = (M >> 4) * (K >> 5) * 32;
    pack_W<<<(nWfrag + 255) / 256, 256, 0, stream>>>(W, Wp, K, M);

    // GEMM: H = x_in @ W  (bf16 WMMA, f32 accumulate)
    if (M % 128 == 0) {
        dim3 g(M / 128, (N + 127) / 128);
        gemm_bf16_wmma<8><<<g, 256, 0, stream>>>(Abf, Wp, Hbuf, N, K, M);
    } else {
        dim3 g(M / 64, (N + 127) / 128);
        gemm_bf16_wmma<4><<<g, 256, 0, stream>>>(Abf, Wp, Hbuf, N, K, M);
    }

    // attention coefficients
    int nwaves = N * Hh;
    attn_coeff<<<(nwaves + 7) / 8, 256, 0, stream>>>(Hbuf, a_s, a_d, alpha_s, alpha_d, N, Hh, D);

    // clear reduction buffers + output accumulator
    hipMemsetAsync(mbuf, 0, (size_t)N * Hh * sizeof(unsigned), stream);
    hipMemsetAsync(sbuf, 0, (size_t)N * Hh * sizeof(float), stream);
    hipMemsetAsync(outBuf, 0, (size_t)N * M * sizeof(float), stream);

    edge_max<<<(Et + 255) / 256, 256, 0, stream>>>(ei, E, N, Hh, alpha_s, alpha_d, mbuf);
    edge_sum<<<(Et + 255) / 256, 256, 0, stream>>>(ei, E, N, Hh, alpha_s, alpha_d, mbuf, sbuf);
    edge_agg<<<(Et + 7) / 8, 256, 0, stream>>>(ei, E, N, Hh, D, alpha_s, alpha_d, mbuf, sbuf, Hbuf, outBuf);

    size_t total = (size_t)N * M;
    bias_act<<<(int)((total + 255) / 256), 256, 0, stream>>>(outBuf, b, total, M, do_elu);
}

extern "C" void kernel_launch(void* const* d_in, const int* in_sizes, int n_in,
                              void* d_out, int out_size, void* d_ws, size_t ws_size,
                              hipStream_t stream)
{
    // setup_inputs order:
    // 0:x 1:edge_index 2:W1 3:as1 4:ad1 5:b1 6:W2 7:as2 8:ad2 9:b2 10:W3 11:as3 12:ad3 13:b3
    const float* x   = (const float*)d_in[0];
    const int*   ei  = (const int*)d_in[1];      // jax default -> int32, [2,E]
    const float* W1  = (const float*)d_in[2];
    const float* as1 = (const float*)d_in[3];
    const float* ad1 = (const float*)d_in[4];
    const float* b1  = (const float*)d_in[5];
    const float* W2  = (const float*)d_in[6];
    const float* as2 = (const float*)d_in[7];
    const float* ad2 = (const float*)d_in[8];
    const float* b2  = (const float*)d_in[9];
    const float* W3  = (const float*)d_in[10];
    const float* as3 = (const float*)d_in[11];
    const float* ad3 = (const float*)d_in[12];
    const float* b3  = (const float*)d_in[13];

    const int F_IN = 256, HID = 128, OUT = 64;
    const int N = in_sizes[0] / F_IN;            // 50000
    const int E = in_sizes[1] / 2;               // 400000

    // workspace carve-up
    char* p = (char*)d_ws;
    size_t off = 0;
    const size_t bigBytes = (size_t)N * 8 * HID * sizeof(float);   // 1024 feats, f32
    float* Hbuf = (float*)(p + off); off = alignUp(off + bigBytes, 256);
    float* Xbuf = (float*)(p + off); off = alignUp(off + bigBytes, 256);
    __bf16* Abf = (__bf16*)(p + off); off = alignUp(off + bigBytes / 2, 256);
    __bf16* Wp  = (__bf16*)(p + off); off = alignUp(off + (size_t)2 * 1024 * 1024, 256);
    float* alpha_s = (float*)(p + off); off = alignUp(off + (size_t)N * 8 * sizeof(float), 256);
    float* alpha_d = (float*)(p + off); off = alignUp(off + (size_t)N * 8 * sizeof(float), 256);
    unsigned* mbuf = (unsigned*)(p + off); off = alignUp(off + (size_t)N * 8 * sizeof(unsigned), 256);
    float* sbuf = (float*)(p + off); off = alignUp(off + (size_t)N * 8 * sizeof(float), 256);
    (void)ws_size; (void)n_in; (void)out_size;

    // Layer 1: 256 -> 8 x 128, ELU
    run_layer(x, W1, as1, ad1, b1, Hbuf, Xbuf, Abf, Wp, alpha_s, alpha_d, mbuf, sbuf,
              ei, N, E, F_IN, 8, HID, 1, stream);
    // Layer 2: 1024 -> 4 x 128, ELU (in/out both through Xbuf: GEMM consumes Xbuf
    // into Abf before the memset+aggregation rewrite it; all ops stream-ordered)
    run_layer(Xbuf, W2, as2, ad2, b2, Hbuf, Xbuf, Abf, Wp, alpha_s, alpha_d, mbuf, sbuf,
              ei, N, E, 8 * HID, 4, HID, 1, stream);
    // Layer 3: 512 -> 1 x 64, no activation, straight to d_out
    run_layer(Xbuf, W3, as3, ad3, b3, Hbuf, (float*)d_out, Abf, Wp, alpha_s, alpha_d, mbuf, sbuf,
              ei, N, E, 4 * HID, 1, OUT, 0, stream);
}
